// MultiheadAttention_45251775430659
// MI455X (gfx1250) — compile-verified, matrix operand order checked by reading, not run on hardware
//
#include <hip/hip_runtime.h>

typedef __bf16 bf16;
typedef __attribute__((ext_vector_type(16))) __bf16 v16bf;
typedef __attribute__((ext_vector_type(8)))  float  v8f;

#define EMBED  512
#define NHEAD  8
#define HD     64
#define NTOK   4096
#define NBATCH 2
#define ROWS   (NBATCH * NTOK)   // 8192

// ---------------------------------------------------------------------------
// WMMA helpers (wave32, 16x16x32 bf16 -> f32)
// ---------------------------------------------------------------------------
__device__ __forceinline__ v8f wmma_bf16(v16bf a, v16bf b, v8f c) {
  return __builtin_amdgcn_wmma_f32_16x16x32_bf16(
      /*neg_a=*/false, a, /*neg_b=*/false, b,
      /*c_mod=*/(short)0, c, /*reuse_a=*/false, /*reuse_b=*/false);
}

__device__ __forceinline__ v8f zero8() {
  v8f z;
#pragma unroll
  for (int i = 0; i < 8; ++i) z[i] = 0.0f;
  return z;
}

// A-fragment (16x32, MxK) from row-major [rows x ld] at (row0, k0), fetched as
// two 128-bit loads per lane. Per ISA 7.12.2: lane l -> row M = l%16;
// lanes 0-15 hold K {0..7, 16..23}, lanes 16-31 hold K {8..15, 24..31}.
// Also serves as the B-fragment of M^T. Requires (ld, k0) multiples of 8 elems.
__device__ __forceinline__ v16bf load_frag(const bf16* __restrict__ base,
                                           int ld, int row0, int k0) {
  const int l = threadIdx.x & 31;
  const bf16* p = base + (size_t)(row0 + (l & 15)) * ld + k0 + ((l & 16) ? 8 : 0);
  union { v16bf v; int4 q[2]; } u;
  u.q[0] = *(const int4*)(p);
  u.q[1] = *(const int4*)(p + 16);
  return u.v;
}

// Async global -> LDS copy of one 16-byte chunk per lane (CDNA5 GV mode).
// VDST = per-lane LDS byte offset (generic LDS pointer low 32 bits),
// VADDR = per-lane 64-bit global address. Tracked by ASYNCcnt.
__device__ __forceinline__ void async_copy_b128(bf16* lds_dst, const bf16* gsrc) {
  uint32_t lo = (uint32_t)(uintptr_t)lds_dst;
  asm volatile("global_load_async_to_lds_b128 %0, %1, off"
               :: "v"(lo), "v"(gsrc) : "memory");
}

// ---------------------------------------------------------------------------
// Kernel 1: x + sinusoidal PE, cast to bf16
// ---------------------------------------------------------------------------
__global__ void pe_add_cast_kernel(const float* __restrict__ x,
                                   bf16* __restrict__ xb) {
  int e = blockIdx.x * blockDim.x + threadIdx.x;
  if (e >= ROWS * EMBED) return;
  int d = e % EMBED;
  int n = (e / EMBED) % NTOK;
  float half2 = (float)((d >> 1) << 1);                      // 2*floor(d/2)
  float inv   = __expf(-logf(10000.0f) * half2 / (float)EMBED);
  float ang   = (float)n * inv;
  float pe    = (d & 1) ? __cosf(ang) : __sinf(ang);
  xb[e] = (bf16)(x[e] + pe);
}

// ---------------------------------------------------------------------------
// Kernel 2: fp32 [K x N] -> bf16 transposed [N x K]
// (B-fragment of W == A-layout load of W^T, so GEMM B loads become b128.)
// ---------------------------------------------------------------------------
__global__ void cast_transpose_kernel(const float* __restrict__ src,
                                      bf16* __restrict__ dstT, int K, int N) {
  int e = blockIdx.x * blockDim.x + threadIdx.x;
  if (e >= K * N) return;
  int k = e / N;
  int n = e % N;
  dstT[(size_t)n * K + k] = (bf16)src[e];
}

// ---------------------------------------------------------------------------
// Kernel 3: QKV GEMM  [8192x512] x [512x1536] via WT [1536x512].
// 8 waves / block; wave w owns a 16(M) x 64(N) tile (A-frag reused 4x).
// Outputs: Q, K as [B,H,N,64]; V transposed as [B,H,64,N] (for P@V B-frags).
// ---------------------------------------------------------------------------
__global__ __launch_bounds__(256) void qkv_gemm_kernel(
    const bf16* __restrict__ X, const bf16* __restrict__ WT,
    bf16* __restrict__ Q, bf16* __restrict__ K, bf16* __restrict__ Vt) {
  const int wave = threadIdx.x >> 5;
  const int row0 = blockIdx.x * 128 + wave * 16;
  const int col0 = blockIdx.y * 64;

  v8f acc[4];
#pragma unroll
  for (int f = 0; f < 4; ++f) acc[f] = zero8();

  for (int k0 = 0; k0 < EMBED; k0 += 32) {
    v16bf a = load_frag(X, EMBED, row0, k0);
#pragma unroll
    for (int f = 0; f < 4; ++f) {
      v16bf b = load_frag(WT, EMBED, col0 + 16 * f, k0);   // B-frag of W
      acc[f] = wmma_bf16(a, b, acc[f]);
    }
  }

  // Epilogue. Blocks never straddle the batch boundary (4096 % 128 == 0).
  const int l    = threadIdx.x & 31;
  const int n    = l & 15;
  const int mo   = (l & 16) ? 8 : 0;
  const int bb   = row0 >> 12;          // row0 / NTOK
  const int tok0 = row0 & (NTOK - 1);
#pragma unroll
  for (int f = 0; f < 4; ++f) {
    const int cbase = col0 + 16 * f;    // 16-col tile sits inside one head
    const int t     = cbase / EMBED;    // 0=Q 1=K 2=V
    const int rem   = cbase % EMBED;
    const int h     = rem / HD;
    const int dbase = rem % HD;
    if (t == 2) {
      // Transposed V: one b128 store of 8 consecutive tokens per lane.
      union { int4 q; bf16 h8[8]; } pk;
#pragma unroll
      for (int r = 0; r < 8; ++r) pk.h8[r] = (bf16)acc[f][r];
      size_t off = ((size_t)(bb * NHEAD + h) * HD + dbase + n) * NTOK + tok0 + mo;
      *(int4*)(Vt + off) = pk.q;
    } else {
      bf16* dst = (t == 0) ? Q : K;
      size_t rowbase = ((size_t)(bb * NHEAD + h) * NTOK + tok0 + mo);
#pragma unroll
      for (int r = 0; r < 8; ++r)
        dst[(rowbase + r) * HD + dbase + n] = (bf16)acc[f][r];
    }
  }
}

// ---------------------------------------------------------------------------
// Kernel 4: flash attention, one wave per (b,h, 16-query tile).
// K/V tiles double-buffered in LDS via global_load_async_to_lds_b128;
// s_wait_asynccnt 16 overlaps next-tile DMA with current-tile WMMAs.
// Online softmax keeps scores in C-fragments; P goes via LDS to A-layout.
// ---------------------------------------------------------------------------
__global__ __launch_bounds__(32) void attn_kernel(
    const bf16* __restrict__ Q, const bf16* __restrict__ K,
    const bf16* __restrict__ Vt, bf16* __restrict__ O /* [8192 x 512] */) {
  __shared__ bf16 Klds[2][32 * 64];   // K tile: 32 keys x 64 dims
  __shared__ bf16 Vlds[2][64 * 32];   // V tile (transposed): 64 dims x 32 keys
  __shared__ bf16 Pt[16 * 32];

  const int bh = blockIdx.x;            // 0..15
  const int b  = bh >> 3;
  const int h  = bh & 7;
  const int q0 = blockIdx.y * 16;

  const bf16* Qh = Q  + (size_t)bh * NTOK * HD;
  const bf16* Kh = K  + (size_t)bh * NTOK * HD;
  const bf16* Vh = Vt + (size_t)bh * HD * NTOK;   // [64 x NTOK]

  const int l  = threadIdx.x & 31;
  const int n  = l & 15;
  const int mo = (l & 16) ? 8 : 0;

  // Stage one K+V tile (8 KB) = 16 async b128 ops (one 16B chunk per lane).
  auto stage_tiles = [&](int key0, bf16* kbuf, bf16* vbuf) {
#pragma unroll
    for (int i = 0; i < 8; ++i) {              // K: 32 rows x 128B (8 chunks/row)
      int chunk = i * 32 + l;                  // 0..255
      int row   = chunk >> 3;
      int c16   = chunk & 7;
      async_copy_b128(kbuf + chunk * 8,
                      Kh + (size_t)(key0 + row) * HD + c16 * 8);
    }
#pragma unroll
    for (int i = 0; i < 8; ++i) {              // V: 64 rows x 64B (4 chunks/row)
      int chunk = i * 32 + l;
      int row   = chunk >> 2;
      int c16   = chunk & 3;
      async_copy_b128(vbuf + chunk * 8,
                      Vh + (size_t)row * NTOK + key0 + c16 * 8);
    }
  };

  // Q fragments for the whole tile (hd=64 -> two k=32 fragments), loaded once.
  v16bf qa0 = load_frag(Qh, HD, q0, 0);
  v16bf qa1 = load_frag(Qh, HD, q0, 32);

  float rmax[8], rsum[8];
  v8f o[4];
#pragma unroll
  for (int r = 0; r < 8; ++r) { rmax[r] = -3.0e38f; rsum[r] = 0.0f; }
#pragma unroll
  for (int f = 0; f < 4; ++f) o[f] = zero8();

  stage_tiles(0, Klds[0], Vlds[0]);            // prime the pipeline

  for (int key0 = 0; key0 < NTOK; key0 += 32) {
    const int cur = (key0 >> 5) & 1;
    if (key0 + 32 < NTOK) {
      stage_tiles(key0 + 32, Klds[cur ^ 1], Vlds[cur ^ 1]);
      // Async loads complete in order: ASYNCcnt <= 16 => current tile landed.
      asm volatile("s_wait_asynccnt 0x10" ::: "memory");
    } else {
      asm volatile("s_wait_asynccnt 0x0" ::: "memory");
    }
    const bf16* kt = Klds[cur];
    const bf16* vt = Vlds[cur];

    // Scores from LDS: B-fragment of K^T == A-layout load of K rows.
    v8f s0 = zero8(), s1 = zero8();
    {
      v16bf kb = load_frag(kt, HD, 0, 0);
      s0 = wmma_bf16(qa0, kb, s0);
      kb = load_frag(kt, HD, 0, 32);
      s0 = wmma_bf16(qa1, kb, s0);
      kb = load_frag(kt, HD, 16, 0);
      s1 = wmma_bf16(qa0, kb, s1);
      kb = load_frag(kt, HD, 16, 32);
      s1 = wmma_bf16(qa1, kb, s1);
    }

    // Online softmax, per register-row r (global row = mo + r).
#pragma unroll
    for (int r = 0; r < 8; ++r) {
      float a0 = s0[r] * 0.125f;        // 1/sqrt(64)
      float a1 = s1[r] * 0.125f;
      float t  = fmaxf(a0, a1);
#pragma unroll
      for (int off = 1; off < 16; off <<= 1)
        t = fmaxf(t, __shfl_xor(t, off, 32));   // reduce within 16-lane group
      float mnew = fmaxf(rmax[r], t);
      float p0 = __expf(a0 - mnew);
      float p1 = __expf(a1 - mnew);
      float ps = p0 + p1;
#pragma unroll
      for (int off = 1; off < 16; off <<= 1)
        ps += __shfl_xor(ps, off, 32);
      float alpha = __expf(rmax[r] - mnew);
      rmax[r] = mnew;
      rsum[r] = rsum[r] * alpha + ps;
#pragma unroll
      for (int f = 0; f < 4; ++f) o[f][r] *= alpha;
      // C-layout -> LDS tile (row = mo+r, cols n and 16+n)
      Pt[(mo + r) * 32 + n]      = (bf16)p0;
      Pt[(mo + r) * 32 + 16 + n] = (bf16)p1;
    }
    __syncthreads();

    // P [16x32] (A-layout from LDS) @ V [32x64] via transposed V tile in LDS.
    v16bf pa = load_frag(Pt, 32, 0, 0);
#pragma unroll
    for (int f = 0; f < 4; ++f) {
      v16bf vb = load_frag(vt, 32, 16 * f, 0);   // B-frag of V
      o[f] = wmma_bf16(pa, vb, o[f]);
    }
    __syncthreads();
  }

  // Normalize and store into [B, N, H*hd] bf16 for the output projection.
#pragma unroll
  for (int r = 0; r < 8; ++r) {
    float inv = 1.0f / rsum[r];
#pragma unroll
    for (int f = 0; f < 4; ++f) o[f][r] *= inv;
  }
#pragma unroll
  for (int f = 0; f < 4; ++f)
#pragma unroll
    for (int r = 0; r < 8; ++r) {
      size_t row = (size_t)b * NTOK + q0 + mo + r;
      O[row * EMBED + h * HD + 16 * f + n] = (bf16)o[f][r];
    }
}

// ---------------------------------------------------------------------------
// Kernel 5: output projection [8192x512] x [512x512] (via WT) + bias -> fp32
// ---------------------------------------------------------------------------
__global__ __launch_bounds__(256) void proj_gemm_kernel(
    const bf16* __restrict__ A, const bf16* __restrict__ WT,
    const float* __restrict__ bias, float* __restrict__ out) {
  const int wave = threadIdx.x >> 5;
  const int row0 = blockIdx.x * 128 + wave * 16;
  const int col0 = blockIdx.y * 64;

  v8f acc[4];
#pragma unroll
  for (int f = 0; f < 4; ++f) acc[f] = zero8();

  for (int k0 = 0; k0 < EMBED; k0 += 32) {
    v16bf a = load_frag(A, EMBED, row0, k0);
#pragma unroll
    for (int f = 0; f < 4; ++f) {
      v16bf b = load_frag(WT, EMBED, col0 + 16 * f, k0);   // B-frag of W
      acc[f] = wmma_bf16(a, b, acc[f]);
    }
  }

  const int l  = threadIdx.x & 31;
  const int n  = l & 15;
  const int mo = (l & 16) ? 8 : 0;
#pragma unroll
  for (int f = 0; f < 4; ++f) {
    int col = col0 + 16 * f + n;
    float bv = bias[col];
#pragma unroll
    for (int r = 0; r < 8; ++r) {
      int row = row0 + mo + r;
      out[(size_t)row * EMBED + col] = acc[f][r] + bv;
    }
  }
}

// ---------------------------------------------------------------------------
extern "C" void kernel_launch(void* const* d_in, const int* in_sizes, int n_in,
                              void* d_out, int out_size, void* d_ws, size_t ws_size,
                              hipStream_t stream) {
  const float* x     = (const float*)d_in[0];
  const float* qkv_w = (const float*)d_in[1];
  const float* lin_w = (const float*)d_in[2];
  const float* lin_b = (const float*)d_in[3];

  char* ws = (char*)d_ws;
  bf16* xb    = (bf16*)ws; ws += (size_t)ROWS * EMBED * sizeof(bf16);
  bf16* wqkvT = (bf16*)ws; ws += (size_t)EMBED * 3 * EMBED * sizeof(bf16);
  bf16* wlinT = (bf16*)ws; ws += (size_t)EMBED * EMBED * sizeof(bf16);
  bf16* Q     = (bf16*)ws; ws += (size_t)ROWS * EMBED * sizeof(bf16);
  bf16* K     = (bf16*)ws; ws += (size_t)ROWS * EMBED * sizeof(bf16);
  bf16* Vt    = (bf16*)ws; ws += (size_t)ROWS * EMBED * sizeof(bf16);
  bf16* AO    = (bf16*)ws; ws += (size_t)ROWS * EMBED * sizeof(bf16);

  pe_add_cast_kernel<<<(ROWS * EMBED + 255) / 256, 256, 0, stream>>>(x, xb);
  cast_transpose_kernel<<<(EMBED * 3 * EMBED + 255) / 256, 256, 0, stream>>>(
      qkv_w, wqkvT, EMBED, 3 * EMBED);
  cast_transpose_kernel<<<(EMBED * EMBED + 255) / 256, 256, 0, stream>>>(
      lin_w, wlinT, EMBED, EMBED);

  qkv_gemm_kernel<<<dim3(ROWS / 128, (3 * EMBED) / 64), 256, 0, stream>>>(
      xb, wqkvT, Q, K, Vt);

  attn_kernel<<<dim3(NBATCH * NHEAD, NTOK / 16), 32, 0, stream>>>(Q, K, Vt, AO);

  proj_gemm_kernel<<<dim3(ROWS / 128, EMBED / 64), 256, 0, stream>>>(
      AO, wlinT, lin_b, (float*)d_out);
}
